// GeodesicSpectralModel_6846177870366
// MI455X (gfx1250) — matrix operand order repaired and checked by reading, not compiled, not run on hardware
//
#include <hip/hip_runtime.h>

typedef __attribute__((ext_vector_type(16))) _Float16 v16h;
typedef __attribute__((ext_vector_type(8)))  float    v8f;
typedef __attribute__((ext_vector_type(8)))  unsigned v8u;

#define DT_STEP 0.1f
#define SHOOT_LR 0.5f

// pack two f32 -> packed f16 dword (v_cvt_pk_rtz_f16_f32)
__device__ __forceinline__ unsigned pk2(float a, float b) {
  return __builtin_bit_cast(unsigned, __builtin_amdgcn_cvt_pkrtz(a, b));
}

__device__ __forceinline__ float fast_tanh(float x) {
#if __has_builtin(__builtin_amdgcn_tanhf)
  return __builtin_amdgcn_tanhf(x);            // v_tanh_f32 (gfx1250 trans pipe)
#else
  float e = __builtin_amdgcn_exp2f(x * 2.8853900817779268f);   // 2*log2(e)
  return 1.0f - 2.0f * __builtin_amdgcn_rcpf(e + 1.0f);
#endif
}

__device__ __forceinline__ v8f wmma16(v16h a, v16h b, v8f c) {
  // D = A(16x32 f16) * B(32x16 f16) + C(16x16 f32)
  return __builtin_amdgcn_wmma_f32_16x16x32_f16(false, a, false, b, (short)0, c,
                                                false, false);
}

// Metric net + analytic d/dc. A1/A2 have the 8 hidden rows DUPLICATED into
// rows 8..15, so every intermediate is valid in all 32 lanes and no cross-lane
// reduction is needed. A2d pre-folds 0.5 * w3[m] * W2^T[m][k] * W1[0][k].
//   z1  = W1^T [c,lam,1]                  (WMMA, rows duplicated)
//   t1  = tanh(z1); s1 = 1 - t1^2
//   z2  = W2^T [t1,1]                     (WMMA A2)
//   d2  = A2d   [s1]                      (WMMA A2d)   == 0.5*w3∘(W2^T (s1∘w1c))
//   z3  = w3.tanh(z2)                      dz3 = sum d2[r]*(1-tanh(z2[r])^2)
// Gamma finalize (shared across tiles) done by the caller.
__device__ __forceinline__ void metric_z3(float cc, float lam, v16h A1, v16h A2,
                                          v16h A2d, const float w3[8],
                                          float& z3o, float& dz3o) {
  const v8f z = {0.f, 0.f, 0.f, 0.f, 0.f, 0.f, 0.f, 0.f};
  v8u b1 = {0u, 0u, 0u, 0u, 0u, 0u, 0u, 0u};
  b1[0] = pk2(cc, lam);     // K=0,1 : inputs
  b1[1] = pk2(1.0f, 0.0f);  // K=2   : bias row
  v8f z1 = wmma16(A1, __builtin_bit_cast(v16h, b1), z);

  float t1[8], s1[8];
#pragma unroll
  for (int r = 0; r < 8; ++r) {
    float t = fast_tanh(z1[r]);
    t1[r] = t;
    s1[r] = fmaf(-t, t, 1.0f);             // sech^2(z1)
  }

  v8u bh = {0u, 0u, 0u, 0u, 0u, 0u, 0u, 0u};
  v8u bd = {0u, 0u, 0u, 0u, 0u, 0u, 0u, 0u};
  bh[0] = pk2(t1[0], t1[1]);   bd[0] = pk2(s1[0], s1[1]);
  bh[1] = pk2(t1[2], t1[3]);   bd[1] = pk2(s1[2], s1[3]);
  bh[2] = pk2(t1[4], t1[5]);   bd[2] = pk2(s1[4], s1[5]);
  bh[3] = pk2(t1[6], t1[7]);   bd[3] = pk2(s1[6], s1[7]);
  bh[4] = pk2(1.0f, 0.0f);     // K=24 bias row (bd has none)
  v8f z2 = wmma16(A2,  __builtin_bit_cast(v16h, bh), z);
  v8f d2 = wmma16(A2d, __builtin_bit_cast(v16h, bd), z);

  float z3 = 0.f, dz3 = 0.f;
#pragma unroll
  for (int r = 0; r < 8; ++r) {
    float t = fast_tanh(z2[r]);            // valid in ALL lanes (rows duplicated)
    z3  = fmaf(w3[r], t, z3);
    dz3 = fmaf(d2[r], fmaf(-t, t, 1.0f), dz3);
  }
  z3o = z3;
  dz3o = dz3;
}

// Flow net: [c, v, lam, A, 1] -> tanh(16) -> dot(16)+bias. 16 hidden rows span
// both lane halves, so this one keeps a genuine cross-half reduction.
__device__ __forceinline__ float flow_dA(float cc, float vv, float lam, float AA,
                                         v16h AF, const float s2[8], float bOut) {
  const v8f z = {0.f, 0.f, 0.f, 0.f, 0.f, 0.f, 0.f, 0.f};
  v8u b = {0u, 0u, 0u, 0u, 0u, 0u, 0u, 0u};
  b[0] = pk2(cc, vv);      // K=0,1
  b[1] = pk2(lam, AA);     // K=2,3
  b[2] = pk2(1.0f, 0.0f);  // K=4 : bias row
  v8f d = wmma16(AF, __builtin_bit_cast(v16h, b), z);
  float p = 0.f;
#pragma unroll
  for (int r = 0; r < 8; ++r) p = fmaf(s2[r], fast_tanh(d[r]), p);
  p += __shfl_xor(p, 16, 32);
  return p + bOut;
}

__global__ __launch_bounds__(256) void geodesic_spectral_kernel(
    const float* __restrict__ c_src_g, const float* __restrict__ c_tgt_g,
    const float* __restrict__ lam_g,   const float* __restrict__ A_src_g,
    const float* __restrict__ mW1, const float* __restrict__ mb1,
    const float* __restrict__ mW2, const float* __restrict__ mb2,
    const float* __restrict__ mW3, const float* __restrict__ mb3,
    const float* __restrict__ sW1, const float* __restrict__ sb1,
    const float* __restrict__ sW2, const float* __restrict__ sb2,
    float* __restrict__ out, int N) {
  const int tid  = blockIdx.x * blockDim.x + threadIdx.x;
  const int lane = threadIdx.x & 31;
  const int waveBase = tid - lane;
  if (waveBase >= N) return;                 // wave-uniform: EXEC stays all-ones
  const int idx = (tid < N) ? tid : (N - 1); // clamp tail loads
  const int m   = lane & 15;
  const int mo  = m & 7;                     // duplicated hidden row index
  const bool hi = lane >= 16;

  // ---- A-matrices (weights^T, f16 16x32 layout; rows M = lane&15) ----
  // Metric nets: hidden rows 0..7 duplicated into rows 8..15.
  v8u a1  = {0u, 0u, 0u, 0u, 0u, 0u, 0u, 0u};
  v8u a2  = {0u, 0u, 0u, 0u, 0u, 0u, 0u, 0u};
  v8u a2d = {0u, 0u, 0u, 0u, 0u, 0u, 0u, 0u};
  v8u af  = {0u, 0u, 0u, 0u, 0u, 0u, 0u, 0u};
  const float w3m = mW3[mo];
  if (!hi) {
    // rows M=m, K=0..7 region
    a1[0] = pk2(mW1[mo],       mW1[8 + mo]);     // W1^T[m][0..1]
    a1[1] = pk2(mb1[mo],       0.0f);            // K=2 bias
    a2[0] = pk2(mW2[0*8 + mo], mW2[1*8 + mo]);   // W2^T[m][0..7]
    a2[1] = pk2(mW2[2*8 + mo], mW2[3*8 + mo]);
    a2[2] = pk2(mW2[4*8 + mo], mW2[5*8 + mo]);
    a2[3] = pk2(mW2[6*8 + mo], mW2[7*8 + mo]);
    // derivative A: 0.5 * w3[m] * W2^T[m][k] * W1[0][k]
    a2d[0] = pk2(0.5f*w3m*mW2[0*8+mo]*mW1[0], 0.5f*w3m*mW2[1*8+mo]*mW1[1]);
    a2d[1] = pk2(0.5f*w3m*mW2[2*8+mo]*mW1[2], 0.5f*w3m*mW2[3*8+mo]*mW1[3]);
    a2d[2] = pk2(0.5f*w3m*mW2[4*8+mo]*mW1[4], 0.5f*w3m*mW2[5*8+mo]*mW1[5]);
    a2d[3] = pk2(0.5f*w3m*mW2[6*8+mo]*mW1[6], 0.5f*w3m*mW2[7*8+mo]*mW1[7]);
    // flow net: rows M=0..15 all used, no duplication
    af[0] = pk2(sW1[0*16 + m], sW1[1*16 + m]);   // sW1^T[m][0..3]
    af[1] = pk2(sW1[2*16 + m], sW1[3*16 + m]);
    af[2] = pk2(sb1[m],        0.0f);            // K=4 bias
  } else {
    a2[4] = pk2(mb2[mo], 0.0f);                  // K=24 bias row, all 16 rows
  }
  const v16h Am1  = __builtin_bit_cast(v16h, a1);
  const v16h Am2  = __builtin_bit_cast(v16h, a2);
  const v16h Am2d = __builtin_bit_cast(v16h, a2d);
  const v16h Afl  = __builtin_bit_cast(v16h, af);

  float w3[8], s2[8];
#pragma unroll
  for (int r = 0; r < 8; ++r) {
    w3[r] = mW3[r];
    s2[r] = sW2[r + (hi ? 8 : 0)];  // lanes>=16 see flow hidden rows 8..15 in D
  }
  const float b3  = mb3[0];
  const float bs2 = sb2[0];

  // ---- per-lane element state ----
  const float cs  = c_src_g[idx];
  const float ct  = c_tgt_g[idx];
  const float lm  = lam_g[idx];
  const float As  = A_src_g[idx];

  const float lmx  = __shfl_xor(lm, 16, 32);
  const float lam0 = hi ? lmx : lm;   // tile0 = elements at lanes 0-15
  const float lam1 = hi ? lm : lmx;   // tile1 = elements at lanes 16-31

  float vShoot = ct - cs;

  // ---- shooting: flow net not needed (only c_final is used) ----
  for (int it = 0; it < 10; ++it) {
    float c = cs, v = vShoot;
    for (int s = 0; s < 10; ++s) {
      const float cx = __shfl_xor(c, 16, 32);
      const float c0 = hi ? cx : c;
      const float c1 = hi ? c : cx;
      float z30, dz30, z31, dz31;
      metric_z3(c0, lam0, Am1, Am2, Am2d, w3, z30, dz30);
      metric_z3(c1, lam1, Am1, Am2, Am2d, w3, z31, dz31);
      // shared gamma finalize: each lane picks its own tile's result
      const float z3  = (hi ? z31 : z30) + b3;
      const float dz3 = hi ? dz31 : dz30;
      const float e   = __builtin_amdgcn_exp2f(z3 * 1.4426950408889634f);
      const float g   = fmaf(__builtin_amdgcn_logf(1.0f + e),
                             0.6931471805599453f, 1e-6f);     // softplus + 1e-6
      const float gam = e * __builtin_amdgcn_rcpf(1.0f + e)   // sigmoid(z3)
                          * dz3 * __builtin_amdgcn_rcpf(g);   // 0.5 folded in A2d
      c = fmaf(v, DT_STEP, c);                 // uses old v
      v = fmaf(-gam * v, v * DT_STEP, v);      // v - gam*v^2*dt
    }
    vShoot = fmaf(-SHOOT_LR, c - ct, vShoot);
  }

  // ---- final integrate with spectral flow ----
  float c = cs, v = vShoot, A = As;
  for (int s = 0; s < 10; ++s) {
    const float cx = __shfl_xor(c, 16, 32);
    const float vx = __shfl_xor(v, 16, 32);
    const float Ax = __shfl_xor(A, 16, 32);
    const float c0 = hi ? cx : c,  c1 = hi ? c : cx;
    const float v0 = hi ? vx : v,  v1 = hi ? v : vx;
    const float A0 = hi ? Ax : A,  A1v = hi ? A : Ax;
    float z30, dz30, z31, dz31;
    metric_z3(c0, lam0, Am1, Am2, Am2d, w3, z30, dz30);
    metric_z3(c1, lam1, Am1, Am2, Am2d, w3, z31, dz31);
    const float dA0  = flow_dA(c0, v0, lam0, A0,  Afl, s2, bs2);
    const float dA1  = flow_dA(c1, v1, lam1, A1v, Afl, s2, bs2);
    const float z3  = (hi ? z31 : z30) + b3;
    const float dz3 = hi ? dz31 : dz30;
    const float e   = __builtin_amdgcn_exp2f(z3 * 1.4426950408889634f);
    const float g   = fmaf(__builtin_amdgcn_logf(1.0f + e),
                           0.6931471805599453f, 1e-6f);
    const float gam = e * __builtin_amdgcn_rcpf(1.0f + e)
                        * dz3 * __builtin_amdgcn_rcpf(g);
    const float dA  = hi ? dA1 : dA0;
    c = fmaf(v, DT_STEP, c);
    v = fmaf(-gam * v, v * DT_STEP, v);
    A = fmaf(dA, DT_STEP, A);
  }

  if (tid < N) out[tid] = A;
}

extern "C" void kernel_launch(void* const* d_in, const int* in_sizes, int n_in,
                              void* d_out, int out_size, void* d_ws, size_t ws_size,
                              hipStream_t stream) {
  (void)n_in; (void)out_size; (void)d_ws; (void)ws_size;
  const float* c_src = (const float*)d_in[0];
  const float* c_tgt = (const float*)d_in[1];
  const float* lam   = (const float*)d_in[2];
  const float* A_src = (const float*)d_in[3];
  const float* mW1 = (const float*)d_in[4];
  const float* mb1 = (const float*)d_in[5];
  const float* mW2 = (const float*)d_in[6];
  const float* mb2 = (const float*)d_in[7];
  const float* mW3 = (const float*)d_in[8];
  const float* mb3 = (const float*)d_in[9];
  const float* sW1 = (const float*)d_in[10];
  const float* sb1 = (const float*)d_in[11];
  const float* sW2 = (const float*)d_in[12];
  const float* sb2 = (const float*)d_in[13];

  const int N = in_sizes[0];
  const int threads = 256;                     // 8 wave32s per block
  const int blocks = (N + threads - 1) / threads;
  geodesic_spectral_kernel<<<blocks, threads, 0, stream>>>(
      c_src, c_tgt, lam, A_src,
      mW1, mb1, mW2, mb2, mW3, mb3,
      sW1, sb1, sW2, sb2,
      (float*)d_out, N);
}
